// ManifoldformerClassifier_987842478685
// MI455X (gfx1250) — matrix operand order repaired and checked by reading, not compiled, not run on hardware
//
#include <hip/hip_runtime.h>
#include <math.h>

// ---------------------------------------------------------------------------
// Manifoldformer forward for MI455X (gfx1250, wave32, WMMA).
// Compute-bound GEMM workload (~0.56 TFLOP); weights/activations L2-resident.
// All contractions on v_wmma_f32_16x16x32_bf16. All f32->bf16 conversion is
// done exactly once per tensor (fused into producers), so the GEMM inner loop
// is pure data movement + WMMA. Full tiles staged via async load-to-LDS
// (ASYNCcnt path, builtin-confirmed on this toolchain), else vector copies.
// Workspace requirement: ~159 MB (see carve in kernel_launch).
// ---------------------------------------------------------------------------

#define B_    16
#define N_    196
#define NP_   224          // attention-map row stride, padded to 7*32
#define C_    768
#define H_    12
#define HD_   64
#define DFF_  3072
#define NCLS_ 1000
#define L_    12
#define BN_RSTD_ 0.9999950000374997f   // 1/sqrt(1+1e-5)

#if defined(__has_builtin)
#if __has_builtin(__builtin_amdgcn_global_load_async_to_lds_b128)
#define HAVE_ASYNC_LDS 1
#endif
#if __has_builtin(__builtin_amdgcn_s_wait_asynccnt)
#define HAVE_WAIT_ASYNC 1
#endif
#if __has_builtin(__builtin_amdgcn_cvt_pk_bf16_f32)
#define HAVE_CVT_PK 1
#endif
#endif

typedef float v8f  __attribute__((ext_vector_type(8)));
typedef __bf16 v16bf  __attribute__((ext_vector_type(16)));
typedef __bf16 bf16x8 __attribute__((ext_vector_type(8)));
typedef __bf16 bf16x2 __attribute__((ext_vector_type(2)));
typedef unsigned short u16x16 __attribute__((ext_vector_type(16)));
typedef unsigned short u16x8v __attribute__((ext_vector_type(8)));

__device__ __forceinline__ unsigned short f2bf_u(float f) {
  unsigned u = __float_as_uint(f);
  return (unsigned short)((u + 0x7FFFu + ((u >> 16) & 1u)) >> 16);  // RNE
}
__device__ __forceinline__ float bf2f(unsigned short u) {
  return __uint_as_float((unsigned)u << 16);
}
__device__ __forceinline__ bf16x8 zero_bf8() {
  u16x8v z = 0;
  return __builtin_bit_cast(bf16x8, z);
}
__device__ __forceinline__ v16bf cat8(bf16x8 lo, bf16x8 hi) {
  return __builtin_shufflevector(lo, hi, 0,1,2,3,4,5,6,7,8,9,10,11,12,13,14,15);
}
__device__ __forceinline__ v8f wmma_bf16(v16bf a, v16bf b, v8f c) {
  return __builtin_amdgcn_wmma_f32_16x16x32_bf16(
      false, a, false, b, (short)0, c, false, false);
}

#ifdef HAVE_ASYNC_LDS
// Builtin prototype (from hipcc diagnostic): param0 = addrspace(1) v4i*,
// param1 = addrspace(3) v4i*, then imm offset, imm cpol.
typedef int gv4i __attribute__((__vector_size__(16)));
__device__ __forceinline__ void async_b128(const void* gsrc, void* ldst) {
  __builtin_amdgcn_global_load_async_to_lds_b128(
      (__attribute__((address_space(1))) gv4i*)(gsrc),
      (__attribute__((address_space(3))) gv4i*)(ldst), 0, 0);
}
__device__ __forceinline__ void wait_async0() {
#ifdef HAVE_WAIT_ASYNC
  __builtin_amdgcn_s_wait_asynccnt(0);
#else
  asm volatile("s_wait_asynccnt 0x0" ::: "memory");
#endif
}
#endif

// ---------------------------------------------------------------------------
// Pack f32 pairs -> bf16 pairs (one dword per thread).
// ---------------------------------------------------------------------------
__global__ __launch_bounds__(256) void cvt_bf16_kernel(
    const float* __restrict__ src, unsigned* __restrict__ dst, int n2) {
  int i = blockIdx.x * 256 + threadIdx.x;
  if (i >= n2) return;
  float a = src[2 * i], b = src[2 * i + 1];
#ifdef HAVE_CVT_PK
  bf16x2 p = __builtin_amdgcn_cvt_pk_bf16_f32(a, b);
  dst[i] = __builtin_bit_cast(unsigned, p);
#else
  dst[i] = (unsigned)f2bf_u(a) | ((unsigned)f2bf_u(b) << 16);
#endif
}

// ---------------------------------------------------------------------------
// Generic GEMM: Out[M,Nv] = A[M,K]_bf16 * W[Nv,K]_bf16^T + bias (+ epilogue)
// EPI: 0 = bias, 1 = bias + exact GELU, 2 = bias + residual add
// OBF: 1 = write bf16 output (OutH), 0 = write f32 output (Out)
// Block 128x128, K-step 32; 8 waves each computing 32x64 (2x4 WMMA tiles).
// ---------------------------------------------------------------------------
template <int EPI, int OBF>
__global__ __launch_bounds__(256) void gemm_wmma_kernel(
    const unsigned short* __restrict__ A, const unsigned short* __restrict__ W,
    const float* __restrict__ bias, const float* __restrict__ Res,
    float* __restrict__ Out, unsigned short* __restrict__ OutH,
    int M, int Nv, int K) {
  __shared__ unsigned short As[128][32];
  __shared__ unsigned short Bs[128][32];

  const int tid = threadIdx.x;
  const int lane = tid & 31, wave = tid >> 5;
  const int laneLow = lane & 15, sel = lane >> 4;
  const int waveM = wave >> 1, waveN = wave & 1;
  const int rowBase = blockIdx.y * 128, colBase = blockIdx.x * 128;
  const bool fullTile = (rowBase + 128 <= M) && (colBase + 128 <= Nv);

  v8f zero8 = {0.f, 0.f, 0.f, 0.f, 0.f, 0.f, 0.f, 0.f};
  v8f acc[2][4];
#pragma unroll
  for (int i = 0; i < 2; ++i)
#pragma unroll
    for (int j = 0; j < 4; ++j) acc[i][j] = zero8;

  const int numK = K >> 5;  // K is a multiple of 32 for every call
  for (int kt = 0; kt < numK; ++kt) {
    const int k0 = kt << 5;
    __syncthreads();
#ifdef HAVE_ASYNC_LDS
    if (fullTile) {
      // Async copy: 2x b128 per thread per matrix, ASYNCcnt-tracked.
#pragma unroll
      for (int t = 0; t < 2; ++t) {
        int chunk = tid + t * 256;
        int row = chunk >> 2, c8 = (chunk & 3) << 3;
        async_b128(A + (size_t)(rowBase + row) * K + k0 + c8, &As[row][c8]);
        async_b128(W + (size_t)(colBase + row) * K + k0 + c8, &Bs[row][c8]);
      }
      wait_async0();
    } else
#endif
    {
#pragma unroll
      for (int t = 0; t < 2; ++t) {
        int chunk = tid + t * 256;
        int row = chunk >> 2, c8 = (chunk & 3) << 3;
        uint4 av = {0u, 0u, 0u, 0u};
        int gr = rowBase + row;
        if (gr < M) av = *reinterpret_cast<const uint4*>(A + (size_t)gr * K + k0 + c8);
        *reinterpret_cast<uint4*>(&As[row][c8]) = av;
        uint4 bv = {0u, 0u, 0u, 0u};
        int gc = colBase + row;
        if (gc < Nv) bv = *reinterpret_cast<const uint4*>(W + (size_t)gc * K + k0 + c8);
        *reinterpret_cast<uint4*>(&Bs[row][c8]) = bv;
      }
    }
    __syncthreads();

    // A fragment (16x32): lanes 0-15 row m K 0..15; lanes 16-31 K 16..31
    v16bf afr[2];
#pragma unroll
    for (int i = 0; i < 2; ++i) {
      int r = waveM * 32 + i * 16 + laneLow;
      int ko = sel * 16;
      bf16x8 lo = *reinterpret_cast<const bf16x8*>(&As[r][ko]);
      bf16x8 hi = *reinterpret_cast<const bf16x8*>(&As[r][ko + 8]);
      afr[i] = cat8(lo, hi);
    }
    // B fragment (32x16): lanes 0-15 col n K 0..15; lanes 16-31 K 16..31
    v16bf bfr[4];
#pragma unroll
    for (int j = 0; j < 4; ++j) {
      int cc = waveN * 64 + j * 16 + laneLow;
      int kb = sel * 16;
      bf16x8 lo = *reinterpret_cast<const bf16x8*>(&Bs[cc][kb]);
      bf16x8 hi = *reinterpret_cast<const bf16x8*>(&Bs[cc][kb + 8]);
      bfr[j] = cat8(lo, hi);
    }
#pragma unroll
    for (int i = 0; i < 2; ++i)
#pragma unroll
      for (int j = 0; j < 4; ++j)
        acc[i][j] = wmma_bf16(afr[i], bfr[j], acc[i][j]);
  }

  // Epilogue. C layout: VGPR e -> row e (lanes 0-15) / row e+8 (lanes 16-31).
#pragma unroll
  for (int i = 0; i < 2; ++i) {
#pragma unroll
    for (int j = 0; j < 4; ++j) {
      int col = colBase + waveN * 64 + j * 16 + laneLow;
      if (col >= Nv) continue;
      float bi = bias ? bias[col] : 0.f;
#pragma unroll
      for (int e = 0; e < 8; ++e) {
        int r = rowBase + waveM * 32 + i * 16 + e + sel * 8;
        if (r >= M) continue;
        float v = acc[i][j][e] + bi;
        if (EPI == 1) v = 0.5f * v * (1.f + erff(v * 0.70710678118654752f));
        if (EPI == 2) v += Res[(size_t)r * Nv + col];
        if (OBF) OutH[(size_t)r * Nv + col] = f2bf_u(v);
        else     Out[(size_t)r * Nv + col] = v;
      }
    }
  }
}

// ---------------------------------------------------------------------------
// LayerNorm over rows (block per row). Writes f32 out and optional bf16 copy.
// ---------------------------------------------------------------------------
__global__ __launch_bounds__(256) void ln_kernel(
    const float* __restrict__ in, float* __restrict__ out,
    unsigned short* __restrict__ outbf,
    const float* __restrict__ g, const float* __restrict__ bta, int len) {
  const int row = blockIdx.x, tid = threadIdx.x;
  const float* x = in + (size_t)row * len;
  float* y = out + (size_t)row * len;
  __shared__ float red[256];

  float s = 0.f;
  for (int c = tid; c < len; c += 256) s += x[c];
  red[tid] = s; __syncthreads();
  for (int o = 128; o > 0; o >>= 1) { if (tid < o) red[tid] += red[tid + o]; __syncthreads(); }
  const float mean = red[0] / (float)len;
  __syncthreads();

  float v = 0.f;
  for (int c = tid; c < len; c += 256) { float d = x[c] - mean; v += d * d; }
  red[tid] = v; __syncthreads();
  for (int o = 128; o > 0; o >>= 1) { if (tid < o) red[tid] += red[tid + o]; __syncthreads(); }
  const float rstd = rsqrtf(red[0] / (float)len + 1e-5f);
  __syncthreads();

  for (int c = tid; c < len; c += 256) {
    float r = (x[c] - mean) * rstd * g[c] + bta[c];
    y[c] = r;
    if (outbf) outbf[(size_t)row * len + c] = f2bf_u(r);
  }
}

// ---------------------------------------------------------------------------
// Attention scores per (b,h) 16x16 WMMA tile of S = q k^T (bf16 inputs).
// Fuses qq/kk norms, Riemannian channel, temp + eval-BN. abn is [B,2H,N,N] f32.
// ---------------------------------------------------------------------------
__global__ __launch_bounds__(32) void attn_scores_kernel(
    const unsigned short* __restrict__ qkvbf, float* __restrict__ abn,
    const float* __restrict__ scale, const float* __restrict__ riem_scale,
    const float* __restrict__ temp, const float* __restrict__ bn_g,
    const float* __restrict__ bn_b, int l) {
  const int lane = threadIdx.x;
  const int t = blockIdx.x, ti = t / 13, tj = t % 13;
  const int h = blockIdx.y, b = blockIdx.z;
  const int laneLow = lane & 15, sel = lane >> 4;

  const unsigned short* qb = qkvbf + (size_t)b * N_ * (3 * C_) + h * HD_;
  const unsigned short* kb = qb + C_;

  __shared__ float qq[16], kk[16];
  {
    const unsigned short* base = sel ? kb : qb;
    int row = (sel ? tj : ti) * 16 + laneLow;
    float s = 0.f;
    if (row < N_) {
      const unsigned short* p = base + (size_t)row * (3 * C_);
#pragma unroll
      for (int d = 0; d < HD_; ++d) { float x = bf2f(p[d]); s += x * x; }
    }
    if (sel) kk[laneLow] = s; else qq[laneLow] = s;
  }
  __syncthreads();

  v8f acc = {0.f, 0.f, 0.f, 0.f, 0.f, 0.f, 0.f, 0.f};
#pragma unroll
  for (int s = 0; s < 2; ++s) {
    bf16x8 alo = zero_bf8(), ahi = zero_bf8();
    {
      int i = ti * 16 + laneLow;
      if (i < N_) {
        const unsigned short* p = qb + (size_t)i * (3 * C_) + s * 32 + sel * 16;
        alo = *reinterpret_cast<const bf16x8*>(p);
        ahi = *reinterpret_cast<const bf16x8*>(p + 8);
      }
    }
    bf16x8 blo = zero_bf8(), bhi = zero_bf8();
    {
      int j = tj * 16 + laneLow;
      if (j < N_) {
        const unsigned short* p = kb + (size_t)j * (3 * C_) + s * 32 + sel * 16;
        blo = *reinterpret_cast<const bf16x8*>(p);
        bhi = *reinterpret_cast<const bf16x8*>(p + 8);
      }
    }
    acc = wmma_bf16(cat8(alo, ahi), cat8(blo, bhi), acc);
  }

  const float sc = scale[l], rsc = riem_scale[l];
  const float ta = temp[l * 24 + h] * BN_RSTD_ * bn_g[l * 24 + h];
  const float ba = bn_b[l * 24 + h];
  const float tr = temp[l * 24 + H_ + h] * BN_RSTD_ * bn_g[l * 24 + H_ + h];
  const float br = bn_b[l * 24 + H_ + h];
  const float kkv = kk[laneLow];
  float* outA = abn + ((size_t)b * 2 * H_ + h) * N_ * N_;
  float* outR = abn + ((size_t)b * 2 * H_ + H_ + h) * N_ * N_;
  const int j = tj * 16 + laneLow;
#pragma unroll
  for (int e = 0; e < 8; ++e) {
    int i = ti * 16 + e + sel * 8;
    if (i < N_ && j < N_) {
      float qkd = acc[e];
      float qqv = qq[e + sel * 8];
      float d = qqv * qqv + kkv * kkv - 2.f * qkd * qkd;
      float r = sqrtf(fmaxf(d, 0.f) + 1e-12f);
      outA[(size_t)i * N_ + j] = (qkd * sc) * ta + ba;
      outR[(size_t)i * N_ + j] = (r * rsc) * tr + br;
    }
  }
}

// ---------------------------------------------------------------------------
// Channel-mixing conv (2H -> H) + row softmax. Output bf16 with rows padded
// to NP_=224 (zero-filled) so attn_av needs no K guards and stays aligned.
// ---------------------------------------------------------------------------
__global__ __launch_bounds__(256) void conv_softmax_kernel(
    const float* __restrict__ abn, const float* __restrict__ conv_w,
    const float* __restrict__ conv_b, unsigned short* __restrict__ amap_bf,
    int l) {
  const int i = blockIdx.x, b = blockIdx.y, tid = threadIdx.x;
  __shared__ float tile[24][N_];
  __shared__ float cw[12][24];
  __shared__ float red[256];

  const float* src = abn + ((size_t)b * 2 * H_) * N_ * N_ + (size_t)i * N_;
  for (int idx = tid; idx < 24 * N_; idx += 256) {
    int c = idx / N_, j = idx % N_;
    tile[c][j] = src[(size_t)c * N_ * N_ + j];
  }
  for (int idx = tid; idx < 288; idx += 256)
    cw[idx / 24][idx % 24] = conv_w[l * 288 + idx];
  __syncthreads();

  unsigned short* dst = amap_bf + ((size_t)b * H_) * N_ * NP_ + (size_t)i * NP_;
  for (int h = 0; h < H_; ++h) {
    const int j = tid;
    float myv = -3.0e38f;
    if (j < N_) {
      float s = conv_b[l * H_ + h];
#pragma unroll
      for (int c = 0; c < 24; ++c) s += cw[h][c] * tile[c][j];
      myv = s;
    }
    red[tid] = myv; __syncthreads();
    for (int o = 128; o > 0; o >>= 1) { if (tid < o) red[tid] = fmaxf(red[tid], red[tid + o]); __syncthreads(); }
    const float m = red[0]; __syncthreads();
    const float e = (j < N_) ? expf(myv - m) : 0.f;
    red[tid] = e; __syncthreads();
    for (int o = 128; o > 0; o >>= 1) { if (tid < o) red[tid] += red[tid + o]; __syncthreads(); }
    const float ssum = red[0]; __syncthreads();
    if (j < NP_)
      dst[(size_t)h * N_ * NP_ + j] = (j < N_) ? f2bf_u(e / ssum) : (unsigned short)0;
  }
}

// ---------------------------------------------------------------------------
// o = softmax(a) * v per (b,h): [196,224pad] x [196,64]; bf16 in, bf16 out.
// ---------------------------------------------------------------------------
__global__ __launch_bounds__(32) void attn_av_kernel(
    const unsigned short* __restrict__ amap_bf,
    const unsigned short* __restrict__ qkvbf,
    unsigned short* __restrict__ obf) {
  const int lane = threadIdx.x;
  const int t = blockIdx.x, ti = t >> 2, td = t & 3;
  const int h = blockIdx.y, b = blockIdx.z;
  const int laneLow = lane & 15, sel = lane >> 4;

  const unsigned short* arow = amap_bf + ((size_t)b * H_ + h) * N_ * NP_;
  const unsigned short* vb = qkvbf + (size_t)b * N_ * (3 * C_) + 2 * C_ + h * HD_;

  v8f acc = {0.f, 0.f, 0.f, 0.f, 0.f, 0.f, 0.f, 0.f};
  for (int s = 0; s < 7; ++s) {
    const int jb = s * 32;
    bf16x8 alo = zero_bf8(), ahi = zero_bf8();
    {
      int i = ti * 16 + laneLow;
      if (i < N_) {  // K direction is pre-padded with zeros, no guards needed
        const unsigned short* p = arow + (size_t)i * NP_ + jb + sel * 16;
        alo = *reinterpret_cast<const bf16x8*>(p);
        ahi = *reinterpret_cast<const bf16x8*>(p + 8);
      }
    }
    u16x16 bu = 0;
    {
      int dcol = td * 16 + laneLow;
      int kb2 = jb + sel * 16;
#pragma unroll
      for (int m = 0; m < 16; ++m) {
        int j = kb2 + m;
        bu[m] = (j < N_) ? vb[(size_t)j * (3 * C_) + dcol] : (unsigned short)0;
      }
    }
    acc = wmma_bf16(cat8(alo, ahi), __builtin_bit_cast(v16bf, bu), acc);
  }

  const int dcol = td * 16 + laneLow;
  unsigned short* ob = obf + (size_t)b * N_ * C_ + h * HD_ + dcol;
#pragma unroll
  for (int e = 0; e < 8; ++e) {
    int i = ti * 16 + e + sel * 8;
    if (i < N_) ob[(size_t)i * C_] = f2bf_u(acc[e]);
  }
}

// ---------------------------------------------------------------------------
__global__ __launch_bounds__(256) void posadd_kernel(
    const float* __restrict__ x, const float* __restrict__ pos,
    float* __restrict__ xbuf, int total) {
  int i = blockIdx.x * 256 + threadIdx.x;
  if (i < total) xbuf[i] = x[i] + pos[i % (N_ * C_)];
}

// ---------------------------------------------------------------------------
// seq_pool: w = softmax_n(h . pool_w + pool_b); pooled = w^T h (bf16 out).
// ---------------------------------------------------------------------------
__global__ __launch_bounds__(256) void seqpool_kernel(
    const float* __restrict__ hb, const float* __restrict__ pool_w,
    const float* __restrict__ pool_b, unsigned short* __restrict__ pooledbf) {
  const int b = blockIdx.x, tid = threadIdx.x;
  __shared__ float w[N_];
  __shared__ float red[256];
  const float* X = hb + (size_t)b * N_ * C_;

  const int n = tid;
  float myv = -3.0e38f;
  if (n < N_) {
    float s = pool_b[0];
    const float* p = X + (size_t)n * C_;
    for (int c = 0; c < C_; ++c) s += p[c] * pool_w[c];
    myv = s;
  }
  red[tid] = myv; __syncthreads();
  for (int o = 128; o > 0; o >>= 1) { if (tid < o) red[tid] = fmaxf(red[tid], red[tid + o]); __syncthreads(); }
  const float m = red[0]; __syncthreads();
  const float e = (n < N_) ? expf(myv - m) : 0.f;
  red[tid] = e; __syncthreads();
  for (int o = 128; o > 0; o >>= 1) { if (tid < o) red[tid] += red[tid + o]; __syncthreads(); }
  const float ssum = red[0]; __syncthreads();
  if (n < N_) w[n] = e / ssum;
  __syncthreads();

  for (int c = tid; c < C_; c += 256) {
    float acc = 0.f;
    for (int nn = 0; nn < N_; ++nn) acc += w[nn] * X[(size_t)nn * C_ + c];
    pooledbf[(size_t)b * C_ + c] = f2bf_u(acc);
  }
}

// ---------------------------------------------------------------------------
static inline dim3 ggrid(int M, int Nv) { return dim3((Nv + 127) / 128, (M + 127) / 128); }

extern "C" void kernel_launch(void* const* d_in, const int* in_sizes, int n_in,
                              void* d_out, int out_size, void* d_ws, size_t ws_size,
                              hipStream_t stream) {
  (void)in_sizes; (void)n_in; (void)out_size; (void)ws_size;
  const float* x       = (const float*)d_in[0];
  const float* pos     = (const float*)d_in[1];
  const float* ln0_g   = (const float*)d_in[2];
  const float* ln0_b   = (const float*)d_in[3];
  const float* qkv_w   = (const float*)d_in[4];
  const float* qkv_b   = (const float*)d_in[5];
  const float* qkvln_g = (const float*)d_in[6];
  const float* qkvln_b = (const float*)d_in[7];
  const float* scale   = (const float*)d_in[8];
  const float* riem    = (const float*)d_in[9];
  const float* temp    = (const float*)d_in[10];
  const float* bn_g    = (const float*)d_in[11];
  const float* bn_b    = (const float*)d_in[12];
  const float* conv_w  = (const float*)d_in[13];
  const float* conv_b  = (const float*)d_in[14];
  const float* proj_w  = (const float*)d_in[15];
  const float* proj_b  = (const float*)d_in[16];
  const float* ln1_g   = (const float*)d_in[17];
  const float* ln1_b   = (const float*)d_in[18];
  const float* ff1_w   = (const float*)d_in[19];
  const float* ff1_b   = (const float*)d_in[20];
  const float* ff2_w   = (const float*)d_in[21];
  const float* ff2_b   = (const float*)d_in[22];
  const float* norm_g  = (const float*)d_in[23];
  const float* norm_b  = (const float*)d_in[24];
  const float* pool_w  = (const float*)d_in[25];
  const float* pool_b  = (const float*)d_in[26];
  const float* fc_w    = (const float*)d_in[27];
  const float* fc_b    = (const float*)d_in[28];

  const size_t SX = (size_t)B_ * N_ * C_;           // 2,408,448
  const size_t SQ = (size_t)B_ * N_ * 3 * C_;       // 7,225,344
  const size_t SA = (size_t)B_ * 2 * H_ * N_ * N_;  // 14,751,744
  const size_t SMP = (size_t)B_ * H_ * N_ * NP_;    // 8,429,568 (padded map)

  // Byte-offset carve (~159 MB total).
  char* base = (char*)d_ws;
  size_t off = 0;
  auto carve = [&](size_t bytes) {
    void* p = base + off;
    off += (bytes + 255) & ~(size_t)255;
    return p;
  };
  float* xbuf            = (float*)carve(SX * 4);          // residual stream
  float* hbuf            = (float*)carve(SX * 4);          // LN f32 scratch
  float* qkvbuf          = (float*)carve(SQ * 4);          // qkv f32 (pre/post LN)
  float* abn             = (float*)carve(SA * 4);          // [B,2H,N,N]; aliased below
  unsigned short* ffbf   = (unsigned short*)abn;           // [B,N,DFF] bf16 (alias, fits)
  unsigned short* amapbf = (unsigned short*)carve(SMP * 2);
  unsigned short* hbf    = (unsigned short*)carve(SX * 2);
  unsigned short* qkvbf  = (unsigned short*)carve(SQ * 2);
  unsigned short* xbf    = (unsigned short*)carve(SX * 2);
  unsigned short* obf    = (unsigned short*)carve(SX * 2);
  unsigned short* poolbf = (unsigned short*)carve((size_t)B_ * C_ * 2);
  unsigned short* wbf    = (unsigned short*)carve((size_t)DFF_ * C_ * 2); // weight scratch

  const int ROWS = B_ * N_;  // 3136
  auto cvt = [&](const float* s, unsigned short* d, size_t n) {
    int n2 = (int)(n / 2);
    cvt_bf16_kernel<<<(n2 + 255) / 256, 256, 0, stream>>>(s, (unsigned*)d, n2);
  };

  posadd_kernel<<<(int)((SX + 255) / 256), 256, 0, stream>>>(x, pos, xbuf, (int)SX);

  for (int l = 0; l < L_; ++l) {
    // --- attention block ---
    ln_kernel<<<ROWS, 256, 0, stream>>>(xbuf, hbuf, hbf,
                                        ln0_g + (size_t)l * C_, ln0_b + (size_t)l * C_, C_);
    cvt(qkv_w + (size_t)l * 3 * C_ * C_, wbf, (size_t)3 * C_ * C_);
    gemm_wmma_kernel<0, 0><<<ggrid(ROWS, 3 * C_), 256, 0, stream>>>(
        hbf, wbf, qkv_b + (size_t)l * 3 * C_, nullptr, qkvbuf, nullptr, ROWS, 3 * C_, C_);
    ln_kernel<<<ROWS, 256, 0, stream>>>(qkvbuf, qkvbuf, qkvbf,
                                        qkvln_g + (size_t)l * 3 * C_, qkvln_b + (size_t)l * 3 * C_, 3 * C_);
    attn_scores_kernel<<<dim3(13 * 13, H_, B_), 32, 0, stream>>>(
        qkvbf, abn, scale, riem, temp, bn_g, bn_b, l);
    conv_softmax_kernel<<<dim3(N_, B_), 256, 0, stream>>>(abn, conv_w, conv_b, amapbf, l);
    attn_av_kernel<<<dim3(13 * 4, H_, B_), 32, 0, stream>>>(amapbf, qkvbf, obf);
    cvt(proj_w + (size_t)l * C_ * C_, wbf, (size_t)C_ * C_);
    gemm_wmma_kernel<2, 0><<<ggrid(ROWS, C_), 256, 0, stream>>>(
        obf, wbf, proj_b + (size_t)l * C_, xbuf, xbuf, nullptr, ROWS, C_, C_);
    // --- FFN block (reference applies LN1 in-place on the residual) ---
    ln_kernel<<<ROWS, 256, 0, stream>>>(xbuf, xbuf, xbf,
                                        ln1_g + (size_t)l * C_, ln1_b + (size_t)l * C_, C_);
    cvt(ff1_w + (size_t)l * DFF_ * C_, wbf, (size_t)DFF_ * C_);
    gemm_wmma_kernel<1, 1><<<ggrid(ROWS, DFF_), 256, 0, stream>>>(
        xbf, wbf, ff1_b + (size_t)l * DFF_, nullptr, nullptr, ffbf, ROWS, DFF_, C_);
    cvt(ff2_w + (size_t)l * C_ * DFF_, wbf, (size_t)C_ * DFF_);
    gemm_wmma_kernel<2, 0><<<ggrid(ROWS, C_), 256, 0, stream>>>(
        ffbf, wbf, ff2_b + (size_t)l * C_, xbuf, xbuf, nullptr, ROWS, C_, DFF_);
  }

  ln_kernel<<<ROWS, 256, 0, stream>>>(xbuf, hbuf, nullptr, norm_g, norm_b, C_);
  seqpool_kernel<<<B_, 256, 0, stream>>>(hbuf, pool_w, pool_b, poolbf);
  cvt(fc_w, wbf, (size_t)NCLS_ * C_);
  gemm_wmma_kernel<0, 0><<<ggrid(B_, NCLS_), 256, 0, stream>>>(
      poolbf, wbf, fc_b, nullptr, (float*)d_out, nullptr, B_, NCLS_, C_);
}
